// InteractionNetwork_5866925326701
// MI455X (gfx1250) — matrix operand order repaired and checked by reading, not compiled, not run on hardware
//
#include <hip/hip_runtime.h>

typedef __attribute__((ext_vector_type(16))) _Float16 v16h;
typedef __attribute__((ext_vector_type(8)))  float    v8f;

#define WAVES_PER_BLOCK 8
#define TILE_LD 64   // LDS activation row stride (f16), K padded to 64

struct F3 { float a, b, c; };

__device__ __forceinline__ v8f wmma_acc(v16h a, v16h b, v8f c) {
  return __builtin_amdgcn_wmma_f32_16x16x32_f16(false, a, false, b, (short)0, c, false, false);
}
__device__ __forceinline__ v8f wmma_z(v16h a, v16h b) {
  v8f z = {0.f, 0.f, 0.f, 0.f, 0.f, 0.f, 0.f, 0.f};  // folds to inline SRC2=0
  return __builtin_amdgcn_wmma_f32_16x16x32_f16(false, a, false, b, (short)0, z, false, false);
}

// B fragment of [W^T ; bias] : B[k][n] = W[n][k] for k<id, Bias[n] at k==id.
// Layout (ISA 7.12.2, f16 B 32x16): elem 0..7 <-> K = c*32 + (lane<16?0:8) + i; elem 8..15 <-> +16;
// column N = ntile*16 + (lane&15).
__device__ __forceinline__ v16h make_bfrag(const float* __restrict__ W,
                                           const float* __restrict__ Bias,
                                           int od, int id, int ntile, int chunk, int lane) {
  int col = ntile * 16 + (lane & 15);
  int kb  = chunk * 32 + ((lane < 16) ? 0 : 8);
  v16h b;
#pragma unroll
  for (int i = 0; i < 16; ++i) {
    int k = kb + (i < 8 ? i : i + 8);
    float w = 0.0f;
    if (col < od) {
      if (k < id)       w = W[col * id + k];
      else if (k == id) w = Bias[col];     // bias row (multiplied by constant-1 feature)
    }
    b[i] = (_Float16)w;
  }
  return b;
}

// A fragment from row-major 16x64 f16 LDS tile (ISA 7.12.2 f16 A 16x32):
// row = lane&15; elem 0..7 <-> K = c*32 + (lane<16?0:8) + i; elem 8..15 <-> +16.
__device__ __forceinline__ v16h afrag_lds(const _Float16* __restrict__ tile, int chunk, int lane) {
  int row = lane & 15;
  const _Float16* rp = tile + row * TILE_LD;
  int b0 = chunk * 32 + ((lane < 16) ? 0 : 8);
  int b1 = chunk * 32 + ((lane < 16) ? 16 : 24);
  v16h a;
#pragma unroll
  for (int i = 0; i < 8; ++i) a[i] = rp[b0 + i];
#pragma unroll
  for (int i = 0; i < 8; ++i) a[8 + i] = rp[b1 + i];
  return a;
}

// Store C tile to LDS with fused ReLU; column 40 is forced to 1.0 (bias feature
// for the next layer's K=40 bias row). Cols 41..47 compute exactly 0.
__device__ __forceinline__ void store_act(_Float16* __restrict__ tile, int ntile,
                                          v8f c, int lane) {
  int col = ntile * 16 + (lane & 15);
  int r0  = (lane < 16) ? 0 : 8;
  bool isBias = (col == 40);
#pragma unroll
  for (int r = 0; r < 8; ++r) {
    _Float16 hv = isBias ? (_Float16)1.0f : (_Float16)fmaxf(c[r], 0.0f);
    tile[(r0 + r) * TILE_LD + col] = hv;
  }
}

struct MlpFrags {
  v16h b1[3];      // layer1: 1 K-chunk (K = nf features + bias row), 3 N-tiles
  v16h b2[2][3];   // layer2: 2 K-chunks (K=40 + bias row at 40), 3 N-tiles
  v16h b3[2];      // layer3: 2 K-chunks, 1 N-tile
};

__device__ __forceinline__ void load_mlp(MlpFrags& F,
    const float* W1, const float* B1, const float* W2, const float* B2,
    const float* W3, const float* B3, int nf, int od, int lane) {
#pragma unroll
  for (int t = 0; t < 3; ++t) {
    F.b1[t] = make_bfrag(W1, B1, 40, nf, t, 0, lane);
#pragma unroll
    for (int k = 0; k < 2; ++k) F.b2[k][t] = make_bfrag(W2, B2, 40, 40, t, k, lane);
  }
#pragma unroll
  for (int k = 0; k < 2; ++k) F.b3[k] = make_bfrag(W3, B3, od, 40, 0, k, lane);
}

// 3-layer MLP for one 16-row tile; all biases folded into K, every WMMA has C=0
// or chains the previous D. Returns final 16x16 C tile (logits).
template <int NF>
__device__ __forceinline__ v8f mlp3_run(const MlpFrags& F, const float (&f)[10],
                                        _Float16* __restrict__ tile, int lane) {
  // Layer-1 A fragment from registers: K<NF -> feature, K==NF -> 1.0 (bias), else 0.
  v16h a;
#pragma unroll
  for (int i = 0; i < 8; ++i) {
    float lo = (i < NF) ? f[i] : (i == NF ? 1.0f : 0.0f);
    int k = 8 + i;
    float hi = (k < NF) ? f[k] : (k == NF ? 1.0f : 0.0f);
    a[i] = (_Float16)((lane < 16) ? lo : hi);
  }
#pragma unroll
  for (int i = 8; i < 16; ++i) a[i] = (_Float16)0.0f;  // K >= 16 unused (NF <= 10)

  // Layer 1: three independent WMMAs back-to-back, then store (hides D->VALU hazard)
  v8f h0 = wmma_z(a, F.b1[0]);
  v8f h1 = wmma_z(a, F.b1[1]);
  v8f h2 = wmma_z(a, F.b1[2]);
  store_act(tile, 0, h0, lane);
  store_act(tile, 1, h1, lane);
  store_act(tile, 2, h2, lane);
  __builtin_amdgcn_wave_barrier();

  // Layer 2: K=41 over 2 chunks; 6 WMMAs interleaved across 3 accumulators
  v16h a20 = afrag_lds(tile, 0, lane);
  v16h a21 = afrag_lds(tile, 1, lane);
  v8f g0 = wmma_z(a20, F.b2[0][0]);
  v8f g1 = wmma_z(a20, F.b2[0][1]);
  v8f g2 = wmma_z(a20, F.b2[0][2]);
  g0 = wmma_acc(a21, F.b2[1][0], g0);
  g1 = wmma_acc(a21, F.b2[1][1], g1);
  g2 = wmma_acc(a21, F.b2[1][2], g2);
  __builtin_amdgcn_wave_barrier();
  store_act(tile, 0, g0, lane);
  store_act(tile, 1, g1, lane);
  store_act(tile, 2, g2, lane);
  __builtin_amdgcn_wave_barrier();

  // Layer 3
  v16h a30 = afrag_lds(tile, 0, lane);
  v16h a31 = afrag_lds(tile, 1, lane);
  v8f o = wmma_z(a30, F.b3[0]);
  o = wmma_acc(a31, F.b3[1], o);
  return o;
}

// ---------------- Kernels ----------------

__global__ void zero_f32_kernel(float* __restrict__ p, long n) {
  long i = (long)blockIdx.x * blockDim.x + threadIdx.x;
  long st = (long)gridDim.x * blockDim.x;
  for (; i < n; i += st) p[i] = 0.0f;
}

// Gather helpers: load one row's features (row = lane&15) for tile base tb.
__device__ __forceinline__ void gather_edge(const int* __restrict__ src,
                                            const int* __restrict__ dst,
                                            const F3* __restrict__ x3,
                                            const float4* __restrict__ ea4,
                                            long tb, int lane, int E, float (&f)[10]) {
  long e = tb + (lane & 15); if (e >= E) e = E - 1;
  int si = src[e], di = dst[e];
  F3 xd = x3[di];
  F3 xs = x3[si];
  float4 av = ea4[e];
  f[0] = xd.a; f[1] = xd.b; f[2] = xd.c;
  f[3] = xs.a; f[4] = xs.b; f[5] = xs.c;
  f[6] = av.x; f[7] = av.y; f[8] = av.z; f[9] = av.w;
}

// R1: [x[dst], x[src], edge_attr] -> e_msg[E,4]; scatter-add into aggr[N,4]
__global__ void __launch_bounds__(256, 1)
r1_kernel(const float* __restrict__ x,
          const int* __restrict__ src, const int* __restrict__ dst,
          const float* __restrict__ ea,
          const float* __restrict__ W1, const float* __restrict__ B1,
          const float* __restrict__ W2, const float* __restrict__ B2,
          const float* __restrict__ W3, const float* __restrict__ B3,
          float* __restrict__ e_msg, float* __restrict__ aggr, int E) {
  __shared__ _Float16 lds[WAVES_PER_BLOCK * 16 * TILE_LD];
  __shared__ float ldsOut[WAVES_PER_BLOCK * 64];
  int lane = threadIdx.x & 31;
  int wv   = threadIdx.x >> 5;
  _Float16* tile = lds + wv * 16 * TILE_LD;
  float* ldsO = ldsOut + wv * 64;
  for (int i = lane; i < 256; i += 32)  // zero K-pad columns 48..63 once
    tile[(i >> 4) * TILE_LD + 48 + (i & 15)] = (_Float16)0.0f;

  MlpFrags F;
  load_mlp(F, W1, B1, W2, B2, W3, B3, /*nf=*/10, /*od=*/4, lane);

  const F3* __restrict__ x3 = (const F3*)x;
  const float4* __restrict__ ea4 = (const float4*)ea;

  long waveId = (long)blockIdx.x * WAVES_PER_BLOCK + wv;
  long step   = (long)gridDim.x * WAVES_PER_BLOCK * 16;

  long tb = waveId * 16;
  if (tb >= E) return;
  float f[10];
  gather_edge(src, dst, x3, ea4, tb, lane, E, f);   // prologue gather

  for (; tb < E; ) {
    // Software pipeline: issue next tile's gathers before computing this tile.
    long tbn = tb + step;
    float fn[10];
    if (tbn < E) gather_edge(src, dst, x3, ea4, tbn, lane, E, fn);

    v8f o = mlp3_run<10>(F, f, tile, lane);

    // Stage D tile -> LDS, then fully coalesced stores + batched atomics
    int colL = lane & 15;
    if (colL < 4) {
      int r0 = (lane < 16) ? 0 : 8;
#pragma unroll
      for (int r = 0; r < 8; ++r) ldsO[(r0 + r) * 4 + colL] = o[r];
    }
    __builtin_amdgcn_wave_barrier();
#pragma unroll
    for (int j = 0; j < 2; ++j) {
      int elem = lane + 32 * j;          // 64 = 16 rows x 4 cols, contiguous
      long ee = tb + (elem >> 2);
      if (ee < E) {
        float v = ldsO[elem];
        e_msg[tb * 4 + elem] = v;
        atomicAdd(&aggr[(long)dst[ee] * 4 + (elem & 3)], v);  // dst row is L0-hot
      }
    }
    __builtin_amdgcn_wave_barrier();

    tb = tbn;
#pragma unroll
    for (int i = 0; i < 10; ++i) f[i] = fn[i];
  }
}

// O: [x, aggr] -> x_tilde[N,3]
__global__ void __launch_bounds__(256, 1)
obj_kernel(const float* __restrict__ x, const float* __restrict__ aggr,
           const float* __restrict__ W1, const float* __restrict__ B1,
           const float* __restrict__ W2, const float* __restrict__ B2,
           const float* __restrict__ W3, const float* __restrict__ B3,
           float* __restrict__ x_t, int N) {
  __shared__ _Float16 lds[WAVES_PER_BLOCK * 16 * TILE_LD];
  __shared__ float ldsOut[WAVES_PER_BLOCK * 64];
  int lane = threadIdx.x & 31;
  int wv   = threadIdx.x >> 5;
  _Float16* tile = lds + wv * 16 * TILE_LD;
  float* ldsO = ldsOut + wv * 64;
  for (int i = lane; i < 256; i += 32)
    tile[(i >> 4) * TILE_LD + 48 + (i & 15)] = (_Float16)0.0f;

  MlpFrags F;
  load_mlp(F, W1, B1, W2, B2, W3, B3, /*nf=*/7, /*od=*/3, lane);

  const F3* __restrict__ x3 = (const F3*)x;
  const float4* __restrict__ ag4 = (const float4*)aggr;

  long waveId = (long)blockIdx.x * WAVES_PER_BLOCK + wv;
  long stride = (long)gridDim.x * WAVES_PER_BLOCK;
  for (long tb = waveId * 16; tb < N; tb += stride * 16) {
    long n = tb + (lane & 15); if (n >= N) n = N - 1;
    F3 xv = x3[n];
    float4 agv = ag4[n];
    float f[10] = {xv.a, xv.b, xv.c, agv.x, agv.y, agv.z, agv.w, 0.f, 0.f, 0.f};

    v8f o = mlp3_run<7>(F, f, tile, lane);

    int colL = lane & 15;
    if (colL < 3) {
      int r0 = (lane < 16) ? 0 : 8;
#pragma unroll
      for (int r = 0; r < 8; ++r) ldsO[(r0 + r) * 3 + colL] = o[r];
    }
    __builtin_amdgcn_wave_barrier();
#pragma unroll
    for (int j = 0; j < 2; ++j) {
      int elem = lane + 32 * j;          // 48 = 16 rows x 3 cols
      long idx = tb * 3 + elem;
      if (elem < 48 && idx < (long)N * 3) x_t[idx] = ldsO[elem];
    }
    __builtin_amdgcn_wave_barrier();
  }
}

__device__ __forceinline__ void gather_edge2(const int* __restrict__ src,
                                             const int* __restrict__ dst,
                                             const F3* __restrict__ xt3,
                                             const float4* __restrict__ em4,
                                             long tb, int lane, int E, float (&f)[10]) {
  long e = tb + (lane & 15); if (e >= E) e = E - 1;
  int si = src[e], di = dst[e];
  F3 xd = xt3[di];
  F3 xs = xt3[si];
  float4 mv = em4[e];
  f[0] = xd.a; f[1] = xd.b; f[2] = xd.c;
  f[3] = xs.a; f[4] = xs.b; f[5] = xs.c;
  f[6] = mv.x; f[7] = mv.y; f[8] = mv.z; f[9] = mv.w;
}

// R2: [x_t[dst], x_t[src], e_msg] -> sigmoid -> out[E]
__global__ void __launch_bounds__(256, 1)
r2_kernel(const float* __restrict__ x_t,
          const int* __restrict__ src, const int* __restrict__ dst,
          const float* __restrict__ e_msg,
          const float* __restrict__ W1, const float* __restrict__ B1,
          const float* __restrict__ W2, const float* __restrict__ B2,
          const float* __restrict__ W3, const float* __restrict__ B3,
          float* __restrict__ out, int E) {
  __shared__ _Float16 lds[WAVES_PER_BLOCK * 16 * TILE_LD];
  __shared__ float ldsOut[WAVES_PER_BLOCK * 64];
  int lane = threadIdx.x & 31;
  int wv   = threadIdx.x >> 5;
  _Float16* tile = lds + wv * 16 * TILE_LD;
  float* ldsO = ldsOut + wv * 64;
  for (int i = lane; i < 256; i += 32)
    tile[(i >> 4) * TILE_LD + 48 + (i & 15)] = (_Float16)0.0f;

  MlpFrags F;
  load_mlp(F, W1, B1, W2, B2, W3, B3, /*nf=*/10, /*od=*/1, lane);

  const F3* __restrict__ xt3 = (const F3*)x_t;
  const float4* __restrict__ em4 = (const float4*)e_msg;

  long waveId = (long)blockIdx.x * WAVES_PER_BLOCK + wv;
  long step   = (long)gridDim.x * WAVES_PER_BLOCK * 16;

  long tb = waveId * 16;
  if (tb >= E) return;
  float f[10];
  gather_edge2(src, dst, xt3, em4, tb, lane, E, f);

  for (; tb < E; ) {
    long tbn = tb + step;
    float fn[10];
    if (tbn < E) gather_edge2(src, dst, xt3, em4, tbn, lane, E, fn);

    v8f o = mlp3_run<10>(F, f, tile, lane);

    int colL = lane & 15;
    if (colL == 0) {
      int r0 = (lane < 16) ? 0 : 8;
#pragma unroll
      for (int r = 0; r < 8; ++r) ldsO[r0 + r] = o[r];
    }
    __builtin_amdgcn_wave_barrier();
    if (lane < 16) {
      long ee = tb + lane;
      if (ee < E) out[ee] = 1.0f / (1.0f + __expf(-ldsO[lane]));
    }
    __builtin_amdgcn_wave_barrier();

    tb = tbn;
#pragma unroll
    for (int i = 0; i < 10; ++i) f[i] = fn[i];
  }
}

extern "C" void kernel_launch(void* const* d_in, const int* in_sizes, int n_in,
                              void* d_out, int out_size, void* d_ws, size_t ws_size,
                              hipStream_t stream) {
  const float* x    = (const float*)d_in[0];
  const int*   ei   = (const int*)d_in[1];     // [2,E]: row0 = src, row1 = dst
  const float* ea   = (const float*)d_in[2];
  const float* r1w1 = (const float*)d_in[3];  const float* r1b1 = (const float*)d_in[4];
  const float* r1w2 = (const float*)d_in[5];  const float* r1b2 = (const float*)d_in[6];
  const float* r1w3 = (const float*)d_in[7];  const float* r1b3 = (const float*)d_in[8];
  const float* ow1  = (const float*)d_in[9];  const float* ob1  = (const float*)d_in[10];
  const float* ow2  = (const float*)d_in[11]; const float* ob2  = (const float*)d_in[12];
  const float* ow3  = (const float*)d_in[13]; const float* ob3  = (const float*)d_in[14];
  const float* r2w1 = (const float*)d_in[15]; const float* r2b1 = (const float*)d_in[16];
  const float* r2w2 = (const float*)d_in[17]; const float* r2b2 = (const float*)d_in[18];
  const float* r2w3 = (const float*)d_in[19]; const float* r2b3 = (const float*)d_in[20];

  int N = in_sizes[0] / 3;
  int E = in_sizes[2] / 4;
  const int* src = ei;
  const int* dst = ei + E;

  float* e_msg = (float*)d_ws;                 // E*4 floats
  float* aggr  = e_msg + (size_t)E * 4;        // N*4 floats
  float* x_t   = aggr  + (size_t)N * 4;        // N*3 floats
  float* out   = (float*)d_out;                // E floats

  {
    long n = (long)N * 4;
    int g = (int)((n + 255) / 256);
    zero_f32_kernel<<<g, 256, 0, stream>>>(aggr, n);
  }
  r1_kernel<<<2048, 256, 0, stream>>>(x, src, dst, ea,
                                      r1w1, r1b1, r1w2, r1b2, r1w3, r1b3,
                                      e_msg, aggr, E);
  obj_kernel<<<512, 256, 0, stream>>>(x, aggr,
                                      ow1, ob1, ow2, ob2, ow3, ob3,
                                      x_t, N);
  r2_kernel<<<2048, 256, 0, stream>>>(x_t, src, dst, e_msg,
                                      r2w1, r2b1, r2w2, r2b2, r2w3, r2b3,
                                      out, E);
}
